// FusionNet_4604204941661
// MI455X (gfx1250) — matrix operand order repaired and checked by reading, not compiled, not run on hardware
//
#include <hip/hip_runtime.h>
#include <cstdint>
#include <cstddef>

// ---------------------------------------------------------------------------
// MI455X / gfx1250 implementation: all GEMMs via v_wmma_f32_16x16x32_bf16,
// adjacency packed to bits once (268MB -> 8MB), flash-style online-softmax
// attention with WMMA accumulation, double-buffered async global->LDS B tiles
// (GLOBAL_LOAD_ASYNC_TO_LDS_B128 + s_wait_asynccnt).
// ---------------------------------------------------------------------------

typedef __bf16 bf16;
typedef __attribute__((ext_vector_type(8)))  bf16  v8bf;
typedef __attribute__((ext_vector_type(16))) bf16  v16bf;
typedef __attribute__((ext_vector_type(8)))  float v8f;

#define N_NODES 4096
#define NF      128

__device__ __forceinline__ v16bf cat16(v8bf lo, v8bf hi) {
  return __builtin_shufflevector(lo, hi, 0,1,2,3,4,5,6,7,8,9,10,11,12,13,14,15);
}
__device__ __forceinline__ v8f wmma_bf16(v16bf a, v16bf b, v8f c) {
  // D = A(16x32) * B(32x16) + C, fp32 accumulate
  return __builtin_amdgcn_wmma_f32_16x16x32_bf16(false, a, false, b, (short)0, c,
                                                 false, false);
}
__device__ __forceinline__ float sigm(float x) { return 1.f / (1.f + __expf(-x)); }

// Async copy of 32 bytes global->LDS (two B128 ops; INST_OFFSET applies to both
// the LDS and global side per CDNA5 ISA 15.18.3). Tracked by ASYNCcnt.
__device__ __forceinline__ void async_copy32(unsigned lds_addr, const void* gaddr) {
  asm volatile("global_load_async_to_lds_b128 %0, %1, off\n\t"
               "global_load_async_to_lds_b128 %0, %1, off offset:16"
               :: "v"(lds_addr), "v"((unsigned long long)(uintptr_t)gaddr)
               : "memory");
}
__device__ __forceinline__ void wait_async0() {
  asm volatile("s_wait_asynccnt 0x0" ::: "memory");
}

// ---------------------------------------------------------------------------
// K0a: pack graph_adjs (4096,4,4096) int32 -> bitmask adjbits[step][row][word]
// This is the only HBM-heavy kernel (268 MB stream): use b128 loads.
// ---------------------------------------------------------------------------
__global__ void k_pack_adj(const int* __restrict__ graph, unsigned* __restrict__ adjbits)
{
  const int t = blockIdx.x * 256 + threadIdx.x;          // 2,097,152 words
  const int w = t & 127;
  const int n = (t >> 7) & 4095;
  const int i = t >> 19;
  const int4* src = (const int4*)(graph + ((size_t)n * 4 + i) * 4096 + w * 32);
  unsigned bits = 0u;
  #pragma unroll
  for (int q = 0; q < 8; q++) {
    const int4 v = src[q];
    bits |= (v.x > 0) ? (1u << (q * 4 + 0)) : 0u;
    bits |= (v.y > 0) ? (1u << (q * 4 + 1)) : 0u;
    bits |= (v.z > 0) ? (1u << (q * 4 + 2)) : 0u;
    bits |= (v.w > 0) ? (1u << (q * 4 + 3)) : 0u;
  }
  adjbits[((size_t)i * 4096 + n) * 128 + w] = bits;
}

// ---------------------------------------------------------------------------
// K0b: weight conversion/transposition to bf16
// ---------------------------------------------------------------------------
__global__ void k_prep_weights(const float* __restrict__ Wg, const float* __restrict__ W_out,
                               const float* __restrict__ conv_w,
                               bf16* __restrict__ WgT, bf16* __restrict__ WoutT,
                               bf16* __restrict__ convT)
{
  const int t = blockIdx.x * 256 + threadIdx.x;
  if (t < 131072) {                       // WgT[g][o][k] = Wg[g][k][o]
    const int g = t >> 15, rem = t & 32767, o = rem >> 8, k = rem & 255;
    WgT[t] = (bf16)Wg[(size_t)(g * 256 + k) * 128 + o];
  } else if (t < 131072 + 16384) {        // WoutT[o][k] = W_out[k][o]
    const int j = t - 131072, o = j >> 7, k = j & 127;
    WoutT[j] = (bf16)W_out[k * 128 + o];
  } else if (t < 131072 + 16384 + 98304) {// convT = conv_w (already [j][o][i*2+d])
    const int j = t - (131072 + 16384);
    convT[j] = (bf16)conv_w[j];
  }
}

// ---------------------------------------------------------------------------
// K0c: init state: hidden = broadcast(h0), cell = actors[:,0,:]
// ---------------------------------------------------------------------------
__global__ void k_init(const float* __restrict__ actors, const float* __restrict__ h0,
                       float* __restrict__ hidden, float* __restrict__ cell)
{
  const int i = blockIdx.x * 256 + threadIdx.x;
  hidden[i] = h0[i & 127];
  cell[i]   = actors[(size_t)(i >> 7) * 640 + (i & 127)];
}

// ---------------------------------------------------------------------------
// K1: Wh = [X | hidden] @ Wg[g]  (4096x256 @ 256x128, bf16 WMMA)
//     outputs: WhT (transposed bf16), e_src/e_dst = Wh @ a halves (fp32)
// block = 256 thr (8 waves), each wave: 16 rows x 128 cols; grid (32, 4 gates)
// ---------------------------------------------------------------------------
__global__ __launch_bounds__(256) void k_gat_gemm(
    const float* __restrict__ actors, const float* __restrict__ hidden,
    const bf16* __restrict__ WgT, const float* __restrict__ ag,
    bf16* __restrict__ WhT, float* __restrict__ e_src, float* __restrict__ e_dst,
    int step)
{
  __shared__ bf16 Ap[128 * 256];
  const int g   = blockIdx.y;
  const int rb  = blockIdx.x * 128;
  const int tid = threadIdx.x;
  for (int idx = tid; idx < 128 * 256; idx += 256) {
    const int r = idx >> 8, k = idx & 255;
    const float v = (k < 128)
        ? actors[(size_t)(rb + r) * 640 + (step + 1) * 128 + k]
        : hidden[(size_t)(rb + r) * 128 + (k - 128)];
    Ap[idx] = (bf16)v;
  }
  __syncthreads();

  const int wave = tid >> 5, lane = tid & 31;
  const int ln = lane & 15, hh = lane >> 4;
  const int row0 = rb + wave * 16;
  const int arow = wave * 16 + ln;
  const bf16* Bg = WgT + (size_t)g * 128 * 256;

  v8f acc[8];
  #pragma unroll
  for (int n = 0; n < 8; n++) acc[n] = (v8f){0.f,0.f,0.f,0.f,0.f,0.f,0.f,0.f};

  #pragma unroll
  for (int kt = 0; kt < 8; kt++) {
    const bf16* ap = &Ap[arow * 256 + kt * 32 + hh * 8];
    const v16bf A = cat16(*(const v8bf*)ap, *(const v8bf*)(ap + 16));
    #pragma unroll
    for (int nt = 0; nt < 8; nt++) {
      const bf16* bp = Bg + (size_t)(nt * 16 + ln) * 256 + kt * 32 + hh * 8;
      const v16bf B = cat16(*(const v8bf*)bp, *(const v8bf*)(bp + 16));
      acc[nt] = wmma_bf16(A, B, acc[nt]);
    }
  }

  // e_src / e_dst = per-row dot with attention vector halves
  float as8[8], ad8[8];
  const float* av = ag + g * 256;
  #pragma unroll
  for (int nt = 0; nt < 8; nt++) { as8[nt] = av[nt*16 + ln]; ad8[nt] = av[128 + nt*16 + ln]; }
  #pragma unroll
  for (int r = 0; r < 8; r++) {
    float ss = 0.f, dd = 0.f;
    #pragma unroll
    for (int nt = 0; nt < 8; nt++) { ss += acc[nt][r] * as8[nt]; dd += acc[nt][r] * ad8[nt]; }
    #pragma unroll
    for (int off = 1; off < 16; off <<= 1) { ss += __shfl_xor(ss, off); dd += __shfl_xor(dd, off); }
    if (ln == 0) {
      const int row = row0 + r + 8 * hh;
      e_src[g * 4096 + row] = ss;
      e_dst[g * 4096 + row] = dd;
    }
  }

  // store WhT bf16 (transposed: [f][row], rows contiguous -> b128 stores)
  #pragma unroll
  for (int nt = 0; nt < 8; nt++) {
    const int col = nt * 16 + ln;
    v8bf pk;
    #pragma unroll
    for (int r = 0; r < 8; r++) pk[r] = (bf16)acc[nt][r];
    *(v8bf*)(WhT + (size_t)g * NF * N_NODES + (size_t)col * N_NODES + row0 + 8 * hh) = pk;
  }
}

// ---------------------------------------------------------------------------
// K2: attention: out = elu(softmax(mask(lrelu(e_src_i+e_dst_j))) @ Wh)
// Flash-style online softmax, P built directly in WMMA A-fragment layout.
// block = 8 waves (each wave owns 16 rows); shared Wh^T 128x32 tile is
// double-buffered in LDS via async global->LDS DMA overlapping the WMMAs.
// ---------------------------------------------------------------------------
__global__ __launch_bounds__(256) void k_gat_attn(
    const bf16* __restrict__ WhT, const float* __restrict__ e_src_g,
    const float* __restrict__ e_dst_g, const unsigned* __restrict__ adjbits,
    float* __restrict__ gat, int step)
{
  __shared__ bf16 Bt[2][128 * 32];
  const int g = blockIdx.y;
  const int tid = threadIdx.x, wave = tid >> 5, lane = tid & 31;
  const int ln = lane & 15, hh = lane >> 4;
  const int row0  = blockIdx.x * 128 + wave * 16;
  const int myrow = row0 + ln;
  const bf16*  Wg  = WhT + (size_t)g * NF * N_NODES;
  const float* ed  = e_dst_g + g * 4096;
  const unsigned* adj = adjbits + (size_t)step * 4096 * 128 + (size_t)myrow * 128;
  const float es = e_src_g[g * 4096 + myrow];

  // per-thread async-copy slot: 32B of the 8KB tile (256 thr * 32B)
  const bf16*    gsrc  = Wg + (size_t)(tid >> 1) * N_NODES + (tid & 1) * 16;
  const unsigned ldsb0 = (unsigned)(uintptr_t)&Bt[0][(tid >> 1) * 32 + (tid & 1) * 16];
  const unsigned ldsb1 = (unsigned)(uintptr_t)&Bt[1][(tid >> 1) * 32 + (tid & 1) * 16];

  float m = -1e30f, l = 0.f;
  v8f acc[8];
  #pragma unroll
  for (int n = 0; n < 8; n++) acc[n] = (v8f){0.f,0.f,0.f,0.f,0.f,0.f,0.f,0.f};

  async_copy32(ldsb0, gsrc);                      // prefetch tile 0

  for (int jt = 0; jt < 128; jt++) {
    const int cur = jt & 1;

    // ---- probabilities for this tile (pure VALU: overlaps in-flight DMA) ----
    const unsigned aw = adj[jt];
    const int kb = hh * 8;
    float v[16];
    {
      const float4* p0 = (const float4*)(ed + jt * 32 + kb);
      const float4 a = p0[0], b = p0[1];
      const float4* p1 = (const float4*)(ed + jt * 32 + kb + 16);
      const float4 c = p1[0], d = p1[1];
      v[0]=a.x; v[1]=a.y; v[2]=a.z; v[3]=a.w; v[4]=b.x; v[5]=b.y; v[6]=b.z; v[7]=b.w;
      v[8]=c.x; v[9]=c.y; v[10]=c.z; v[11]=c.w; v[12]=d.x; v[13]=d.y; v[14]=d.z; v[15]=d.w;
    }
    float tmax = -1e30f;
    #pragma unroll
    for (int e = 0; e < 16; e++) {
      const int kidx = kb + (e < 8 ? e : e + 8);
      float x = es + v[e];
      x = x > 0.f ? x : 0.2f * x;               // leaky_relu (alpha 0.2)
      x = ((aw >> kidx) & 1u) ? x : -1e9f;      // mask (NEG_INF)
      v[e] = x;
      tmax = fmaxf(tmax, x);
    }
    tmax = fmaxf(tmax, __shfl_xor(tmax, 16));
    const float mnew = fmaxf(m, tmax);
    const float s = __expf(m - mnew);
    m = mnew;

    float psum = 0.f;
    v16bf P;
    #pragma unroll
    for (int e = 0; e < 16; e++) {
      const float p = __expf(v[e] - mnew);
      psum += p;
      P[e] = (bf16)p;
    }
    psum += __shfl_xor(psum, 16);
    l = l * s + psum;

    if (__ballot(s != 1.0f)) {                  // rescale only when max moved
      float sr[8];
      #pragma unroll
      for (int r = 0; r < 8; r++) sr[r] = __shfl(s, r + 8 * hh);
      #pragma unroll
      for (int nt = 0; nt < 8; nt++)
        #pragma unroll
        for (int r = 0; r < 8; r++) acc[nt][r] *= sr[r];
    }

    // ---- tile jt ready: own DMA done, barrier publishes all waves' parts ----
    wait_async0();
    __syncthreads();
    // reads of buffer cur^1 finished before this barrier -> safe to refill it
    if (jt + 1 < 128)
      async_copy32(cur ? ldsb0 : ldsb1, gsrc + (size_t)(jt + 1) * 32);

    #pragma unroll
    for (int nt = 0; nt < 8; nt++) {
      const bf16* bp = &Bt[cur][(nt * 16 + ln) * 32 + hh * 8];
      const v16bf B = cat16(*(const v8bf*)bp, *(const v8bf*)(bp + 16));
      acc[nt] = wmma_bf16(P, B, acc[nt]);
    }
  }

  const float linv = 1.f / l;
  float lr8[8];
  #pragma unroll
  for (int r = 0; r < 8; r++) lr8[r] = __shfl(linv, r + 8 * hh);
  #pragma unroll
  for (int nt = 0; nt < 8; nt++) {
    const int col = nt * 16 + ln;
    #pragma unroll
    for (int r = 0; r < 8; r++) {
      float o = acc[nt][r] * lr8[r];
      o = o > 0.f ? o : __expf(o) - 1.f;        // ELU
      gat[(size_t)g * 524288 + (size_t)(row0 + r + 8 * hh) * 128 + col] = o;
    }
  }
}

// ---------------------------------------------------------------------------
// K3: LSTM pointwise update
// ---------------------------------------------------------------------------
__global__ void k_lstm(const float* __restrict__ gat, float* __restrict__ cell,
                       float* __restrict__ hidden)
{
  const int i = blockIdx.x * 256 + threadIdx.x;
  const float fg = sigm(gat[i]);
  const float ig = sigm(gat[524288 + i]);
  const float cc = tanhf(gat[2 * 524288 + i]);
  const float og = sigm(gat[3 * 524288 + i]);
  const float c = cell[i] * fg + ig * cc;
  cell[i]   = c;
  hidden[i] = tanhf(c) * og;
}

// ---------------------------------------------------------------------------
// K4: out = sigmoid(hidden @ W_out + b_out) -> x0[n][o][step] (bf16)
// ---------------------------------------------------------------------------
__global__ __launch_bounds__(256) void k_out_gemm(
    const float* __restrict__ hidden, const bf16* __restrict__ WoutT,
    const float* __restrict__ b_out, bf16* __restrict__ x0, int step)
{
  __shared__ bf16 Ap[128 * 128];
  const int rb  = blockIdx.x * 128;
  const int tid = threadIdx.x;
  for (int idx = tid; idx < 128 * 128; idx += 256)
    Ap[idx] = (bf16)hidden[(size_t)(rb + (idx >> 7)) * 128 + (idx & 127)];
  __syncthreads();

  const int wave = tid >> 5, lane = tid & 31;
  const int ln = lane & 15, hh = lane >> 4;
  const int row0 = rb + wave * 16;
  const int arow = wave * 16 + ln;

  v8f acc[8];
  #pragma unroll
  for (int n = 0; n < 8; n++) acc[n] = (v8f){0.f,0.f,0.f,0.f,0.f,0.f,0.f,0.f};

  #pragma unroll
  for (int kt = 0; kt < 4; kt++) {
    const bf16* ap = &Ap[arow * 128 + kt * 32 + hh * 8];
    const v16bf A = cat16(*(const v8bf*)ap, *(const v8bf*)(ap + 16));
    #pragma unroll
    for (int nt = 0; nt < 8; nt++) {
      const bf16* bp = WoutT + (size_t)(nt * 16 + ln) * 128 + kt * 32 + hh * 8;
      const v16bf B = cat16(*(const v8bf*)bp, *(const v8bf*)(bp + 16));
      acc[nt] = wmma_bf16(A, B, acc[nt]);
    }
  }

  #pragma unroll
  for (int nt = 0; nt < 8; nt++) {
    const int col = nt * 16 + ln;
    const float b = b_out[col];
    #pragma unroll
    for (int r = 0; r < 8; r++) {
      const int row = row0 + r + 8 * hh;
      x0[(size_t)row * 512 + col * 4 + step] = (bf16)sigm(acc[nt][r] + b);
    }
  }
}

// ---------------------------------------------------------------------------
// K5: conv1d (kernel=2) as K=256 GEMM: out[n,o,t] = sum_{i,d} x[n,i,t+d] W[o,i,d]
// ---------------------------------------------------------------------------
__global__ __launch_bounds__(256) void k_conv(
    const bf16* __restrict__ xin, int Tin, const bf16* __restrict__ Wc,
    const float* __restrict__ bias, bf16* __restrict__ xout, int Tout,
    float* __restrict__ fout)
{
  __shared__ bf16 Ap[128 * 256];
  const int t   = blockIdx.y;
  const int rb  = blockIdx.x * 128;
  const int tid = threadIdx.x;
  for (int idx = tid; idx < 128 * 128; idx += 256) {     // gather taps
    const int r = idx >> 7, i = idx & 127;
    const bf16* s = xin + (size_t)(rb + r) * 128 * Tin + i * Tin + t;
    Ap[r * 256 + i * 2]     = s[0];
    Ap[r * 256 + i * 2 + 1] = s[1];
  }
  __syncthreads();

  const int wave = tid >> 5, lane = tid & 31;
  const int ln = lane & 15, hh = lane >> 4;
  const int row0 = rb + wave * 16;
  const int arow = wave * 16 + ln;

  v8f acc[8];
  #pragma unroll
  for (int n = 0; n < 8; n++) acc[n] = (v8f){0.f,0.f,0.f,0.f,0.f,0.f,0.f,0.f};

  #pragma unroll
  for (int kt = 0; kt < 8; kt++) {
    const bf16* ap = &Ap[arow * 256 + kt * 32 + hh * 8];
    const v16bf A = cat16(*(const v8bf*)ap, *(const v8bf*)(ap + 16));
    #pragma unroll
    for (int nt = 0; nt < 8; nt++) {
      const bf16* bp = Wc + (size_t)(nt * 16 + ln) * 256 + kt * 32 + hh * 8;
      const v16bf B = cat16(*(const v8bf*)bp, *(const v8bf*)(bp + 16));
      acc[nt] = wmma_bf16(A, B, acc[nt]);
    }
  }

  #pragma unroll
  for (int nt = 0; nt < 8; nt++) {
    const int col = nt * 16 + ln;
    const float b = bias[col];
    #pragma unroll
    for (int r = 0; r < 8; r++) {
      const int row = row0 + r + 8 * hh;
      const float v = acc[nt][r] + b;
      if (fout) fout[(size_t)row * 128 + col] = v;
      else      xout[(size_t)row * 128 * Tout + col * Tout + t] = (bf16)v;
    }
  }
}

// ---------------------------------------------------------------------------
extern "C" void kernel_launch(void* const* d_in, const int* in_sizes, int n_in,
                              void* d_out, int out_size, void* d_ws, size_t ws_size,
                              hipStream_t stream)
{
  const float* actors = (const float*)d_in[0];
  const int*   graph  = (const int*)d_in[1];
  const float* Wg     = (const float*)d_in[2];
  const float* ag     = (const float*)d_in[3];
  const float* W_out  = (const float*)d_in[4];
  const float* b_out  = (const float*)d_in[5];
  const float* h0     = (const float*)d_in[6];
  const float* conv_w = (const float*)d_in[7];
  const float* conv_b = (const float*)d_in[8];
  float* out = (float*)d_out;

  char* ws = (char*)d_ws;
  size_t off = 0;
  auto take = [&](size_t bytes) -> char* {
    char* p = ws + off;
    off = (off + bytes + 255) & ~(size_t)255;
    return p;
  };
  unsigned* adjbits = (unsigned*)take(4ull * 4096 * 128 * 4);   // 8 MB bitmasks
  bf16*  WgT   = (bf16*)take(4ull * 128 * 256 * 2);
  bf16*  WoutT = (bf16*)take(128ull * 128 * 2);
  bf16*  convT = (bf16*)take(3ull * 128 * 256 * 2);
  bf16*  WhT   = (bf16*)take(4ull * 128 * 4096 * 2);            // per-step, 4 gates
  float* es    = (float*)take(4ull * 4096 * 4);
  float* ed    = (float*)take(4ull * 4096 * 4);
  float* gat   = (float*)take(4ull * 4096 * 128 * 4);
  float* cell  = (float*)take(4096ull * 128 * 4);
  float* hid   = (float*)take(4096ull * 128 * 4);
  bf16*  x0    = (bf16*)take(4096ull * 128 * 4 * 2);
  bf16*  x1    = (bf16*)take(4096ull * 128 * 3 * 2);
  bf16*  x2    = (bf16*)take(4096ull * 128 * 2 * 2);

  k_pack_adj<<<8192, 256, 0, stream>>>(graph, adjbits);
  k_prep_weights<<<960, 256, 0, stream>>>(Wg, W_out, conv_w, WgT, WoutT, convT);
  k_init<<<2048, 256, 0, stream>>>(actors, h0, hid, cell);

  for (int step = 0; step < 4; step++) {
    k_gat_gemm<<<dim3(32, 4), 256, 0, stream>>>(actors, hid, WgT, ag, WhT, es, ed, step);
    k_gat_attn<<<dim3(32, 4), 256, 0, stream>>>(WhT, es, ed, adjbits, gat, step);
    k_lstm<<<2048, 256, 0, stream>>>(gat, cell, hid);
    k_out_gemm<<<32, 256, 0, stream>>>(hid, WoutT, b_out, x0, step);
  }

  k_conv<<<dim3(32, 3), 256, 0, stream>>>(x0, 4, convT,         conv_b,       x1, 3, nullptr);
  k_conv<<<dim3(32, 2), 256, 0, stream>>>(x1, 3, convT + 32768, conv_b + 128, x2, 2, nullptr);
  k_conv<<<dim3(32, 1), 256, 0, stream>>>(x2, 2, convT + 65536, conv_b + 256, nullptr, 1, out);
}
